// ResNet_15504831938852
// MI455X (gfx1250) — matrix operand order repaired and checked by reading, not compiled, not run on hardware
//
#include <hip/hip_runtime.h>
#include <hip/hip_bf16.h>

// ---------------------------------------------------------------------------
// CDNA5 (gfx1250) sparse ResNet basic block via V_WMMA_F32_16X16X32_BF16,
// with weight fragments staged into LDS through the async global->LDS path.
// ---------------------------------------------------------------------------

typedef __attribute__((ext_vector_type(16))) __bf16 v16bf;  // 32 B: one WMMA A/B fragment per lane
typedef __attribute__((ext_vector_type(8)))  __bf16 v8bf;   // 16 B chunk
typedef __attribute__((ext_vector_type(8)))  float  v8f;    // 16x16 f32 C/D fragment per lane

#define BN_EPS 1e-4f

union AFrag {
    struct { v8bf lo, hi; } p;   // lo = K chunk0, hi = K chunk1 (per ISA 16-bit A layout)
    v16bf v;
};

struct __align__(8) bf4 { __bf16 x, y, z, w; };

// ---------------------------------------------------------------------------
// Pre-swizzle weights (9,64,64) fp32 -> bf16 in exact B-fragment order.
// Fragment (s,t): covers K = s*32..s*32+31 (K = tap*64 + cin), N = t*16..t*16+15.
// Lane layout (16-bit B, 32x16): lanes 0-15 hold Klocal 0-7,16-23, lanes 16-31 hold 8-15,24-31;
// element e of the lane's v16bf is Klocal = kbase + e (e<8) or 16 + kbase + e-8, column N = t*16 + (lane&15).
// ---------------------------------------------------------------------------
__global__ void prep_w_kernel(const float* __restrict__ w, __bf16* __restrict__ wb) {
    int g = blockIdx.x * blockDim.x + threadIdx.x;
    if (g >= 18 * 4 * 32) return;
    int lane = g & 31;
    int frag = g >> 5;
    int s = frag >> 2, t = frag & 3;
    int n  = t * 16 + (lane & 15);
    int kb = (lane & 16) ? 8 : 0;
    __bf16* dst = wb + ((size_t)frag * 32 + lane) * 16;
#pragma unroll
    for (int e = 0; e < 16; ++e) {
        int K = s * 32 + (e < 8 ? kb + e : 16 + kb + (e - 8));
        int ktap = K >> 6, c = K & 63;
        dst[e] = (__bf16)w[(ktap * 64 + c) * 64 + n];
    }
}

// fp32 -> bf16 cast of features, plus a zeroed pad row at row N (rulebook pad index).
__global__ void cast_feats_kernel(const float* __restrict__ f, __bf16* __restrict__ fb,
                                  int n4, int ntot4) {
    int g = blockIdx.x * blockDim.x + threadIdx.x;
    if (g >= ntot4) return;
    float4 v = make_float4(0.f, 0.f, 0.f, 0.f);
    if (g < n4) v = ((const float4*)f)[g];
    bf4 o = { (__bf16)v.x, (__bf16)v.y, (__bf16)v.z, (__bf16)v.w };
    ((bf4*)fb)[g] = o;
}

// ---------------------------------------------------------------------------
// Gathered GEMM: one wave computes a 16-row x 64-col output tile.
// All 72 KB of B-fragments are staged once per block into LDS with
// GLOBAL_LOAD_ASYNC_TO_LDS_B128 (ASYNCcnt), then each K-step reads B via
// ds_load while the VMEM pipe does the data-dependent A gathers.
// K loop: 18 steps of 16x16x32 bf16 WMMA x 4 column tiles = 72 v_wmma / tile.
// Also emits deterministic per-block channel sum / sum^2 partials for BN.
// ---------------------------------------------------------------------------
__global__ void __launch_bounds__(256)
subm_conv_wmma(const __bf16* __restrict__ fb,   // (N+1) x 64 bf16, row N == 0
               const __bf16* __restrict__ wb,   // 18*4 B-fragments (72 KB)
               const int*    __restrict__ nbr,  // N x 9 (pad index == N)
               float* __restrict__ y,           // N x 64 fp32 conv output
               float* __restrict__ partial,     // nblk x 128 (sum[64], sumsq[64])
               int N) {
    __shared__ __bf16 wsm[18 * 4 * 32 * 16];    // 72 KB staged weight fragments
    __shared__ float  lsum[512];
    __shared__ float  lsq[512];

    const int tid  = threadIdx.x;
    const int lane = tid & 31;
    const int wave = tid >> 5;

    // ---- async-stage the 72 KB weight block into LDS (4608 x 16 B chunks) ----
    {
        unsigned lbase = (unsigned)(size_t)(&wsm[0]);   // low 32 bits = LDS byte offset
#pragma unroll
        for (int i = 0; i < 18; ++i) {
            int chunk = tid + i * 256;                  // 18 * 256 = 4608 chunks
            unsigned long long ga =
                (unsigned long long)(size_t)wb + (unsigned long long)chunk * 16ull;
            unsigned la = lbase + (unsigned)chunk * 16u;
            asm volatile("global_load_async_to_lds_b128 %0, %1, off"
                         :: "v"(la), "v"(ga) : "memory");
        }
        asm volatile("s_wait_asynccnt 0x0" ::: "memory");
    }
    __syncthreads();

    const int tileBase = (blockIdx.x * 8 + wave) * 16;
    const int mrow = lane & 15;
    const int m = tileBase + mrow;

    // neighbor rulebook row for this voxel (both lane halves need the same row)
    int nb[9];
    if (m < N) {
        const int* p = nbr + (size_t)m * 9;
#pragma unroll
        for (int k = 0; k < 9; ++k) nb[k] = p[k];
    } else {
#pragma unroll
        for (int k = 0; k < 9; ++k) nb[k] = N;   // pad row -> zeros, keeps EXEC uniform
    }

    const int koff = (lane & 16) ? 8 : 0;        // ISA 16-bit A layout: lane-half K offset

    v8f acc[4] = { v8f{}, v8f{}, v8f{}, v8f{} };

#pragma unroll
    for (int s = 0; s < 18; ++s) {
        const int tap = s >> 1;
        const int co  = ((s & 1) << 5) + koff;   // channel offset of K chunk0
        const __bf16* pa = fb + (size_t)nb[tap] * 64 + co;
        AFrag a;
        a.p.lo = *(const v8bf*)(pa);             // Klocal koff..koff+7
        a.p.hi = *(const v8bf*)(pa + 16);        // Klocal koff+16..koff+23
        const v16bf* pb = (const v16bf*)(wsm + ((size_t)(s * 4) * 32 + lane) * 16);
#pragma unroll
        for (int t = 0; t < 4; ++t) {
            v16bf b = pb[t * 32];                // fragments are 32 lanes * 32 B apart
            acc[t] = __builtin_amdgcn_wmma_f32_16x16x32_bf16(
                         false, a.v, false, b, (short)0, acc[t], false, false);
        }
    }

    // store results + build per-wave channel partial sums
    const int mb = (lane & 16) ? 8 : 0;          // C/D layout: lane-half M offset
#pragma unroll
    for (int t = 0; t < 4; ++t) {
        float s1 = 0.f, q1 = 0.f;
#pragma unroll
        for (int j = 0; j < 8; ++j) {
            int row = tileBase + mb + j;
            float x = acc[t][j];
            if (row < N) {
                y[(size_t)row * 64 + t * 16 + mrow] = x;
                s1 += x;
                q1 += x * x;
            }
        }
        float s1o = __shfl_xor(s1, 16, 32);
        float q1o = __shfl_xor(q1, 16, 32);
        if (lane < 16) {
            lsum[wave * 64 + t * 16 + mrow] = s1 + s1o;
            lsq [wave * 64 + t * 16 + mrow] = q1 + q1o;
        }
    }
    __syncthreads();
    if (tid < 64) {
        float s = 0.f, q = 0.f;
#pragma unroll
        for (int w8 = 0; w8 < 8; ++w8) {
            s += lsum[w8 * 64 + tid];
            q += lsq [w8 * 64 + tid];
        }
        partial[(size_t)blockIdx.x * 128 + tid]      = s;
        partial[(size_t)blockIdx.x * 128 + 64 + tid] = q;
    }
}

// Fold BN into per-channel a,b:  a = gamma*rsqrt(var+eps); b = beta - mean*a.
__global__ void bn_coef_kernel(const float* __restrict__ partial, int nblk,
                               const float* __restrict__ gamma, const float* __restrict__ beta,
                               float* __restrict__ ab, float invN) {
    int c = threadIdx.x;
    if (c >= 64) return;
    float s = 0.f, q = 0.f;
    for (int b = 0; b < nblk; ++b) {
        s += partial[(size_t)b * 128 + c];
        q += partial[(size_t)b * 128 + 64 + c];
    }
    float mean = s * invN;
    float var  = q * invN - mean * mean;
    float a    = gamma[c] * rsqrtf(var + BN_EPS);
    ab[c]      = a;
    ab[64 + c] = beta[c] - mean * a;
}

// x1 = bf16(relu(y1*a + b)) fused, written over the feats-bf16 buffer (rows 0..N-1).
__global__ void bn_relu_cast_kernel(const float* __restrict__ y, const float* __restrict__ ab,
                                    __bf16* __restrict__ xb, int n4) {
    int g = blockIdx.x * blockDim.x + threadIdx.x;
    if (g >= n4) return;
    float4 v = ((const float4*)y)[g];
    int cg = g & 15;
    float4 a = ((const float4*)ab)[cg];
    float4 b = ((const float4*)(ab + 64))[cg];
    float r0 = fmaxf(v.x * a.x + b.x, 0.f);
    float r1 = fmaxf(v.y * a.y + b.y, 0.f);
    float r2 = fmaxf(v.z * a.z + b.z, 0.f);
    float r3 = fmaxf(v.w * a.w + b.w, 0.f);
    bf4 o = { (__bf16)r0, (__bf16)r1, (__bf16)r2, (__bf16)r3 };
    ((bf4*)xb)[g] = o;
}

// out = relu(y2*a + b + feats), in place on d_out (y2 lives in d_out).
__global__ void residual_kernel(float* __restrict__ out, const float* __restrict__ feats,
                                const float* __restrict__ ab, int n4) {
    int g = blockIdx.x * blockDim.x + threadIdx.x;
    if (g >= n4) return;
    float4 v = ((const float4*)out)[g];
    float4 f = ((const float4*)feats)[g];
    int cg = g & 15;
    float4 a = ((const float4*)ab)[cg];
    float4 b = ((const float4*)(ab + 64))[cg];
    float4 r;
    r.x = fmaxf(v.x * a.x + b.x + f.x, 0.f);
    r.y = fmaxf(v.y * a.y + b.y + f.y, 0.f);
    r.z = fmaxf(v.z * a.z + b.z + f.z, 0.f);
    r.w = fmaxf(v.w * a.w + b.w + f.w, 0.f);
    ((float4*)out)[g] = r;
}

// ---------------------------------------------------------------------------
extern "C" void kernel_launch(void* const* d_in, const int* in_sizes, int n_in,
                              void* d_out, int out_size, void* d_ws, size_t ws_size,
                              hipStream_t stream) {
    const float* feats = (const float*)d_in[0];
    const int*   nbr   = (const int*)d_in[1];
    const float* w1    = (const float*)d_in[2];
    const float* g1    = (const float*)d_in[3];
    const float* b1    = (const float*)d_in[4];
    const float* w2    = (const float*)d_in[5];
    const float* g2    = (const float*)d_in[6];
    const float* b2    = (const float*)d_in[7];
    float*       out   = (float*)d_out;
    (void)n_in; (void)out_size; (void)ws_size;

    const int N        = in_sizes[0] / 64;
    const int numTiles = (N + 15) / 16;
    const int nblk     = (numTiles + 7) / 8;

    // workspace carve-up (256 B aligned)
    char*  ws  = (char*)d_ws;
    size_t off = 0;
    auto carve = [&](size_t bytes) -> void* {
        void* p = ws + off;
        off = (off + bytes + 255) & ~(size_t)255;
        return p;
    };
    __bf16* wb1     = (__bf16*)carve(18 * 4 * 32 * 16 * sizeof(__bf16));
    __bf16* wb2     = (__bf16*)carve(18 * 4 * 32 * 16 * sizeof(__bf16));
    float*  ab1     = (float*)carve(128 * sizeof(float));
    float*  ab2     = (float*)carve(128 * sizeof(float));
    float*  partial = (float*)carve((size_t)nblk * 128 * sizeof(float));
    __bf16* fb      = (__bf16*)carve((size_t)(N + 1) * 64 * sizeof(__bf16));
    float*  y1      = (float*)carve((size_t)N * 64 * sizeof(float));

    const int n4 = N * 16, ntot4 = (N + 1) * 16;
    const float invN = 1.0f / (float)N;

    prep_w_kernel<<<(18 * 4 * 32 + 255) / 256, 256, 0, stream>>>(w1, wb1);
    prep_w_kernel<<<(18 * 4 * 32 + 255) / 256, 256, 0, stream>>>(w2, wb2);
    cast_feats_kernel<<<(ntot4 + 255) / 256, 256, 0, stream>>>(feats, fb, n4, ntot4);

    subm_conv_wmma<<<nblk, 256, 0, stream>>>(fb, wb1, nbr, y1, partial, N);
    bn_coef_kernel<<<1, 64, 0, stream>>>(partial, nblk, g1, b1, ab1, invN);
    bn_relu_cast_kernel<<<(n4 + 255) / 256, 256, 0, stream>>>(y1, ab1, fb, n4);

    subm_conv_wmma<<<nblk, 256, 0, stream>>>(fb, wb2, nbr, out, partial, N);
    bn_coef_kernel<<<1, 64, 0, stream>>>(partial, nblk, g2, b2, ab2, invN);
    residual_kernel<<<(n4 + 255) / 256, 256, 0, stream>>>(out, feats, ab2, n4);
}
